// HyperAttention_66288525246897
// MI455X (gfx1250) — matrix operand order repaired and checked by reading, not compiled, not run on hardware
//
#include <hip/hip_runtime.h>
#include <hip/hip_bf16.h>
#include <math.h>
#include <stdint.h>

#define B_     2
#define H_     16
#define N_     4096
#define D_     64
#define NPROJ  8
#define BLK    256
#define NB     (N_/BLK)
#define SAMPLE 256
#define SCALE  0.125f
#define BH     (B_*H_)
#define BHN    (BH*N_)

typedef __attribute__((ext_vector_type(16))) __bf16 v16bf;
typedef __attribute__((ext_vector_type(8)))  float  v8f;

union ABf { v16bf v; __bf16 e[16]; };
union Cf  { v8f   v; float  e[8];  };

// ---------------------------------------------------------------------------
// Kernel 1: LSH hash (Gray-code permuted bucket id) for Q (y=0) and K (y=1)
// ---------------------------------------------------------------------------
__global__ __launch_bounds__(256) void ha_hash(const float* __restrict__ q,
                                               const float* __restrict__ k,
                                               const float* __restrict__ pd,
                                               int* __restrict__ hq,
                                               int* __restrict__ hk) {
  __shared__ float pdir[D_][NPROJ];
  const int tid = threadIdx.x;
  for (int i = tid; i < D_ * NPROJ; i += 256) pdir[i >> 3][i & 7] = pd[i];
  __syncthreads();

  const int g = blockIdx.x * 256 + tid;
  const float* xp = ((blockIdx.y == 0) ? q : k) + (size_t)g * D_;

  float acc[NPROJ];
#pragma unroll
  for (int r = 0; r < NPROJ; ++r) acc[r] = 0.f;
#pragma unroll 4
  for (int c = 0; c < 16; ++c) {
    float4 x4 = *(const float4*)(xp + c * 4);
#pragma unroll
    for (int r = 0; r < NPROJ; ++r) {
      acc[r] = fmaf(x4.x, pdir[c * 4 + 0][r], acc[r]);
      acc[r] = fmaf(x4.y, pdir[c * 4 + 1][r], acc[r]);
      acc[r] = fmaf(x4.z, pdir[c * 4 + 2][r], acc[r]);
      acc[r] = fmaf(x4.w, pdir[c * 4 + 3][r], acc[r]);
    }
  }
  int bin = 0;
#pragma unroll
  for (int r = 0; r < NPROJ; ++r) if (acc[r] > 0.f) bin |= (1 << r);

  // Gray-code permutation: a_k built by a = concat(a, flip(a)+len)
  int add = 0, x = bin;
#pragma unroll
  for (int kk = NPROJ; kk > 1; --kk) {
    int half = 1 << (kk - 1);
    if (x >= half) { add += half; x = (1 << kk) - 1 - x; }
  }
  ((blockIdx.y == 0) ? hq : hk)[g] = add + x;
}

// ---------------------------------------------------------------------------
// Kernel 2: stable sort by bucket == bitonic sort of (bucket<<12 | index)
// ---------------------------------------------------------------------------
__global__ __launch_bounds__(256) void ha_sort(const int* __restrict__ hq,
                                               const int* __restrict__ hk,
                                               int* __restrict__ qidx,
                                               int* __restrict__ kidx) {
  __shared__ unsigned sk[N_];
  const int bh = blockIdx.x;
  const int* h = (blockIdx.y == 0) ? hq : hk;
  int* o = (blockIdx.y == 0) ? qidx : kidx;

  for (int i = threadIdx.x; i < N_; i += 256)
    sk[i] = (((unsigned)h[(size_t)bh * N_ + i]) << 12) | (unsigned)i;
  __syncthreads();

  for (int kk = 2; kk <= N_; kk <<= 1) {
    for (int j = kk >> 1; j > 0; j >>= 1) {
      for (int i = threadIdx.x; i < N_; i += 256) {
        int ixj = i ^ j;
        if (ixj > i) {
          unsigned a = sk[i], b = sk[ixj];
          bool up = ((i & kk) == 0);
          if ((a > b) == up) { sk[i] = b; sk[ixj] = a; }
        }
      }
      __syncthreads();
    }
  }
  for (int i = threadIdx.x; i < N_; i += 256)
    o[(size_t)bh * N_ + i] = (int)(sk[i] & 4095u);
}

// ---------------------------------------------------------------------------
// Kernel 3: pre-pack sampled K (row-major) / V (transposed) as bf16 so the
// residual kernel can pull them with contiguous async global->LDS copies.
// ---------------------------------------------------------------------------
__global__ __launch_bounds__(256) void ha_pack(const float* __restrict__ k,
                                               const float* __restrict__ v,
                                               const int* __restrict__ samp,
                                               __bf16* __restrict__ kpack,
                                               __bf16* __restrict__ vtpack) {
  const int bh = blockIdx.x;
  const size_t bhBase = (size_t)bh * N_;
  __bf16* kp = kpack + (size_t)bh * SAMPLE * D_;     // [row][d]
  __bf16* vp = vtpack + (size_t)bh * SAMPLE * D_;    // [d][row]
  for (int t = threadIdx.x; t < SAMPLE * 16; t += 256) {
    int row = t >> 4;
    int ch  = (t & 15) * 4;
    int src = samp[bh * SAMPLE + row];
    float4 kf = *(const float4*)(k + (bhBase + src) * D_ + ch);
    float4 vf = *(const float4*)(v + (bhBase + src) * D_ + ch);
    kp[row * D_ + ch + 0] = (__bf16)kf.x; kp[row * D_ + ch + 1] = (__bf16)kf.y;
    kp[row * D_ + ch + 2] = (__bf16)kf.z; kp[row * D_ + ch + 3] = (__bf16)kf.w;
    vp[(ch + 0) * SAMPLE + row] = (__bf16)vf.x;
    vp[(ch + 1) * SAMPLE + row] = (__bf16)vf.y;
    vp[(ch + 2) * SAMPLE + row] = (__bf16)vf.z;
    vp[(ch + 3) * SAMPLE + row] = (__bf16)vf.w;
  }
}

// ---------------------------------------------------------------------------
// Kernel 4/5: 256x256 softmax attention block via WMMA bf16, 8 waves/block.
//  RESIDUAL=false: sorted-block attention, scatter out via q_idx, write LSE.
//  RESIDUAL=true : sampled-set attention (async LDS fill from packed bf16),
//                  fuse LSE-sigmoid combine into d_out.
// ---------------------------------------------------------------------------
template <bool RESIDUAL>
__global__ __launch_bounds__(256) void ha_attn(const float* __restrict__ q,
                                               const float* __restrict__ k,
                                               const float* __restrict__ v,
                                               const int* __restrict__ qidx,
                                               const int* __restrict__ kidx,
                                               const __bf16* __restrict__ kpack,
                                               const __bf16* __restrict__ vtpack,
                                               float* __restrict__ out,
                                               float* __restrict__ lsebuf) {
  __shared__ __bf16 Ks[BLK][D_];     // keys,   row-major [key][d]
  __shared__ __bf16 Vt[D_][BLK];     // values, transposed [d][key]
  __shared__ __bf16 Pb[8][16][BLK];  // per-wave probability strip

  const int bh   = blockIdx.x / NB;
  const int blk  = blockIdx.x % NB;
  const int tid  = threadIdx.x;
  const int wave = tid >> 5;
  const int lane = tid & 31;
  const int m    = lane & 15;   // col (B/C) or row (A) within 16-grid
  const int hi   = lane >> 4;

  const size_t bhBase = (size_t)bh * N_;

  if (RESIDUAL) {
    // contiguous bf16 payload: async global -> LDS, 16B per lane-op
    uint32_t ldsK = (uint32_t)(uintptr_t)(&Ks[0][0]);
    uint32_t ldsV = (uint32_t)(uintptr_t)(&Vt[0][0]);
    uint64_t gK = (uint64_t)(uintptr_t)(kpack + (size_t)bh * SAMPLE * D_);
    uint64_t gV = (uint64_t)(uintptr_t)(vtpack + (size_t)bh * SAMPLE * D_);
    for (int t = tid; t < (SAMPLE * D_ * 2) / 16; t += 256) {  // 2048 chunks
      uint32_t lk = ldsK + t * 16;
      uint64_t ak = gK + (uint64_t)t * 16;
      asm volatile("global_load_async_to_lds_b128 %0, %1, off"
                   :: "v"(lk), "v"(ak) : "memory");
      uint32_t lv = ldsV + t * 16;
      uint64_t av = gV + (uint64_t)t * 16;
      asm volatile("global_load_async_to_lds_b128 %0, %1, off"
                   :: "v"(lv), "v"(av) : "memory");
    }
    asm volatile("s_wait_asynccnt 0x0" ::: "memory");
  } else {
    // gathered f32 rows -> convert -> LDS (conversion forces the VGPR path)
    for (int t = tid; t < BLK * 16; t += 256) {
      int row = t >> 4;
      int ch  = (t & 15) * 4;
      int src = kidx[bhBase + (size_t)blk * BLK + row];
      float4 kf = *(const float4*)(k + (bhBase + src) * D_ + ch);
      float4 vf = *(const float4*)(v + (bhBase + src) * D_ + ch);
      Ks[row][ch + 0] = (__bf16)kf.x; Ks[row][ch + 1] = (__bf16)kf.y;
      Ks[row][ch + 2] = (__bf16)kf.z; Ks[row][ch + 3] = (__bf16)kf.w;
      Vt[ch + 0][row] = (__bf16)vf.x; Vt[ch + 1][row] = (__bf16)vf.y;
      Vt[ch + 2][row] = (__bf16)vf.z; Vt[ch + 3][row] = (__bf16)vf.w;
    }
  }
  __syncthreads();

  for (int rt = wave; rt < 16; rt += 8) {
    const int rowbase = rt * 16;

    // ---- A fragments (Q rows), 16x32 bf16 layout ----
    int qrow = RESIDUAL ? (blk * BLK + rowbase + m)
                        : qidx[bhBase + (size_t)blk * BLK + rowbase + m];
    const float* qp = q + (bhBase + qrow) * D_;
    ABf a0, a1;
#pragma unroll
    for (int vv = 0; vv < 8; ++vv) {
      int kk = 16 * hi + 2 * vv;
      float2 f0 = *(const float2*)(qp + kk);
      float2 f1 = *(const float2*)(qp + 32 + kk);
      a0.e[2 * vv] = (__bf16)f0.x; a0.e[2 * vv + 1] = (__bf16)f0.y;
      a1.e[2 * vv] = (__bf16)f1.x; a1.e[2 * vv + 1] = (__bf16)f1.y;
    }

    // ---- S = Q * K^T : 16 col-tiles x 2 WMMA (K=32 each) ----
    Cf acc[16];
#pragma unroll
    for (int ct = 0; ct < 16; ++ct) {
      const __bf16* kr = &Ks[ct * 16 + m][hi * 16];   // B[k][n] = K[n][k]
      v16bf b0 = *(const v16bf*)kr;
      v16bf b1 = *(const v16bf*)(kr + 32);
      v8f c = {};
      c = __builtin_amdgcn_wmma_f32_16x16x32_bf16(false, a0.v, false, b0,
                                                  (short)0, c, false, false);
      c = __builtin_amdgcn_wmma_f32_16x16x32_bf16(false, a1.v, false, b1,
                                                  (short)0, c, false, false);
      acc[ct].v = c;
    }

    // ---- softmax stats per row (row = j + 8*hi), scaled scores ----
    float lse[8];
#pragma unroll
    for (int j = 0; j < 8; ++j) {
      float mx = -3.0e38f;
#pragma unroll
      for (int ct = 0; ct < 16; ++ct) mx = fmaxf(mx, acc[ct].e[j]);
#pragma unroll
      for (int off = 1; off < 16; off <<= 1)
        mx = fmaxf(mx, __shfl_xor(mx, off, 32));
      mx *= SCALE;
      float s = 0.f;
#pragma unroll
      for (int ct = 0; ct < 16; ++ct)
        s += __expf(acc[ct].e[j] * SCALE - mx);
#pragma unroll
      for (int off = 1; off < 16; off <<= 1)
        s += __shfl_xor(s, off, 32);
      lse[j] = mx + __logf(s);
    }

    // ---- P = exp(s - lse) -> per-wave LDS strip (A-fragment source) ----
#pragma unroll
    for (int ct = 0; ct < 16; ++ct) {
#pragma unroll
      for (int j = 0; j < 8; ++j) {
        Pb[wave][j + 8 * hi][ct * 16 + m] =
            (__bf16)__expf(acc[ct].e[j] * SCALE - lse[j]);
      }
    }

    // ---- O = P * V : 8 K-chunks x 4 col-tiles ----
    Cf oacc[4];
#pragma unroll
    for (int ct = 0; ct < 4; ++ct) { v8f z = {}; oacc[ct].v = z; }
#pragma unroll
    for (int kc = 0; kc < 8; ++kc) {
      ABf ap;
      const __bf16* pr = &Pb[wave][m][kc * 32];
#pragma unroll
      for (int vv = 0; vv < 8; ++vv) {
        int kk = 16 * hi + 2 * vv;
        ap.e[2 * vv]     = pr[kk];
        ap.e[2 * vv + 1] = pr[kk + 1];
      }
#pragma unroll
      for (int ct = 0; ct < 4; ++ct) {
        const __bf16* vr = &Vt[ct * 16 + m][kc * 32 + hi * 16]; // B[k][n]=Vt[n][k]
        v16bf bv = *(const v16bf*)vr;
        oacc[ct].v = __builtin_amdgcn_wmma_f32_16x16x32_bf16(
            false, ap.v, false, bv, (short)0, oacc[ct].v, false, false);
      }
    }

    // ---- epilogue ----
#pragma unroll
    for (int j = 0; j < 8; ++j) {
      int r = rowbase + j + 8 * hi;
      if (!RESIDUAL) {
        int orow = qidx[bhBase + (size_t)blk * BLK + r];
        float* op = out + (bhBase + orow) * D_;
#pragma unroll
        for (int ct = 0; ct < 4; ++ct) op[ct * 16 + m] = oacc[ct].e[j];
        if (m == 0) lsebuf[bhBase + orow] = lse[j];
      } else {
        int row = blk * BLK + r;
        float l1 = lsebuf[bhBase + row];
        float l2 = lse[j] + 2.7725887222397811f;  // + log(N/SAMPLE)=log(16)
        float cc = 1.f / (1.f + __expf(l2 - l1)); // sigmoid(l1-l2)
        float* op = out + (bhBase + row) * D_;
#pragma unroll
        for (int ct = 0; ct < 4; ++ct) {
          float a1v = op[ct * 16 + m];
          op[ct * 16 + m] = cc * a1v + (1.f - cc) * oacc[ct].e[j];
        }
      }
    }
  }
}

// ---------------------------------------------------------------------------
extern "C" void kernel_launch(void* const* d_in, const int* in_sizes, int n_in,
                              void* d_out, int out_size, void* d_ws,
                              size_t ws_size, hipStream_t stream) {
  (void)in_sizes; (void)n_in; (void)out_size; (void)ws_size;
  const float* q  = (const float*)d_in[0];
  const float* k  = (const float*)d_in[1];
  const float* v  = (const float*)d_in[2];
  const float* pd = (const float*)d_in[3];
  const int* samp = (const int*)d_in[4];
  float* out = (float*)d_out;

  int* hq    = (int*)d_ws;
  int* hk    = hq + BHN;
  int* qidx  = hk + BHN;
  int* kidx  = qidx + BHN;
  float* lse = (float*)(kidx + BHN);
  __bf16* kpack  = (__bf16*)(lse + BHN);
  __bf16* vtpack = kpack + (size_t)BH * SAMPLE * D_;

  ha_hash<<<dim3(BHN / 256, 2), 256, 0, stream>>>(q, k, pd, hq, hk);
  ha_sort<<<dim3(BH, 2), 256, 0, stream>>>(hq, hk, qidx, kidx);
  ha_pack<<<dim3(BH), 256, 0, stream>>>(k, v, samp, kpack, vtpack);
  ha_attn<false><<<dim3(BH * NB), 256, 0, stream>>>(q, k, v, qidx, kidx,
                                                    kpack, vtpack, out, lse);
  ha_attn<true><<<dim3(BH * NB), 256, 0, stream>>>(q, k, v, qidx, kidx,
                                                   kpack, vtpack, out, lse);
}
